// GNN_6442450944201
// MI455X (gfx1250) — matrix operand order repaired and checked by reading, not compiled, not run on hardware
//
#include <hip/hip_runtime.h>
#include <hip/hip_bf16.h>

// ---------------------------------------------------------------------------
// 3-layer GCN for MI455X (gfx1250, wave32).
//   layer: H = X @ W      (bf16 WMMA, f32 accum; W pre-packed in WMMA lane order)
//          AGG[dst] += H[src] * dinv[src]*dinv[dst]   (native f32 atomics, L2-resident)
//          OUT = relu(AGG + H*dinv^2 + b)             (self-loop fused here)
// ---------------------------------------------------------------------------

typedef __attribute__((ext_vector_type(16))) __bf16 v16bf;
typedef __attribute__((ext_vector_type(8)))  float  v8f;

static __device__ __forceinline__ __bf16 f2bf(float f) {
  union { float f; unsigned u; } v; v.f = f;
  unsigned r = v.u + 0x7FFFu + ((v.u >> 16) & 1u);  // round-to-nearest-even
  unsigned short hs = (unsigned short)(r >> 16);
  __bf16 b;
  __builtin_memcpy(&b, &hs, sizeof(b));
  return b;
}

// Guaranteed-native f32 atomic add (no-return form, STOREcnt-tracked).
// Avoids any chance of the compiler expanding atomicrmw fadd into a CAS loop.
static __device__ __forceinline__ void atomic_add_f32(float* p, float v) {
  asm volatile("global_atomic_add_f32 %0, %1, off" : : "v"(p), "v"(v) : "memory");
}

__global__ void gcn_zero4(float4* __restrict__ p, long long n4) {
  long long i = (long long)blockIdx.x * blockDim.x + threadIdx.x;
  if (i < n4) p[i] = make_float4(0.f, 0.f, 0.f, 0.f);
}

__global__ void gcn_degree(const int* __restrict__ dst, float* __restrict__ deg, int e) {
  int i = blockIdx.x * blockDim.x + threadIdx.x;
  if (i < e) atomic_add_f32(&deg[dst[i]], 1.0f);
}

__global__ void gcn_rsqrt(float* __restrict__ dinv, int n) {
  int i = blockIdx.x * blockDim.x + threadIdx.x;
  if (i < n) dinv[i] = rsqrtf(dinv[i] + 1.0f);  // +1 = self loop; always > 0
}

// Pre-pack W (f32, K=128 x cols row-major) into bf16 WMMA B-operand lane order:
//   Bp[((ks32*ntiles + tn)*32 + lane)*16 + j] = bf16( W[k, n] )
//   k = ks32*32 + (lane>>4)*8 + (j&7) + (j>>3)*16 ;  n = tn*16 + (lane&15)
// so each (wave-lane, k-step, tile) fetch in the GEMM is 16 contiguous bf16 (32B).
__global__ void gcn_pack_w(const float* __restrict__ Wm, __bf16* __restrict__ Bp,
                           int ntiles) {
  const int cols = ntiles * 16;
  const int total = cols * 128;
  int idx = blockIdx.x * blockDim.x + threadIdx.x;
  if (idx >= total) return;
  const int j    = idx & 15;
  const int lane = (idx >> 4) & 31;
  const int rest = idx >> 9;                // ks32*ntiles + tn
  const int tn   = rest & (ntiles - 1);     // ntiles is 8 or 4
  const int ks32 = rest >> (ntiles == 8 ? 3 : 2);
  const int k = ks32 * 32 + (lane >> 4) * 8 + (j & 7) + (j >> 3) * 16;
  const int n = tn * 16 + (lane & 15);
  Bp[idx] = f2bf(Wm[(size_t)k * cols + n]);
}

// One wave computes a full 16 x (NT*16) output stripe: A (16x32 bf16 tile) is
// loaded once per k-step with four b128 loads and reused across all NT column
// tiles; B comes from the packed buffer as one 32B vector load per (k,tile).
// Layouts per ISA 7.12.2 (wave32):
//   A: lane<16 -> row M=lane,   K = {ks+0..7, ks+16..23}
//      lane>=16 -> row M=lane-16, K = {ks+8..15, ks+24..31}
//   B: lanes 0-15 K={ks+0..7, ks+16..23}, lanes 16-31 K={ks+8..15, ks+24..31}, column N = lane&15
//   C/D: VGPR r of lane l -> (M = r + 8*(l>>4), N = l&15)
template <int NT>
__global__ void gcn_gemm_wmma(const float* __restrict__ X,
                              const __bf16* __restrict__ Bp,
                              float* __restrict__ H,
                              int mtiles) {
  const int cols = NT * 16;
  const int wave = (int)(((long long)blockIdx.x * blockDim.x + threadIdx.x) >> 5);
  const int lane = threadIdx.x & 31;
  if (wave >= mtiles) return;

  const int half = lane >> 4;   // 0 or 1
  const int lm   = lane & 15;
  const float* xrow = X + (size_t)(wave * 16 + lm) * 128;

  v8f acc[NT];
#pragma unroll
  for (int t = 0; t < NT; ++t) acc[t] = (v8f){};

#pragma unroll
  for (int ks32 = 0; ks32 < 4; ++ks32) {
    const int ks = ks32 * 32;
    const float4 a0 = *(const float4*)(xrow + ks + half * 8);
    const float4 a1 = *(const float4*)(xrow + ks + half * 8 + 4);
    const float4 a2 = *(const float4*)(xrow + ks + 16 + half * 8);
    const float4 a3 = *(const float4*)(xrow + ks + 16 + half * 8 + 4);
    v16bf a;
    a[0]  = f2bf(a0.x); a[1]  = f2bf(a0.y); a[2]  = f2bf(a0.z); a[3]  = f2bf(a0.w);
    a[4]  = f2bf(a1.x); a[5]  = f2bf(a1.y); a[6]  = f2bf(a1.z); a[7]  = f2bf(a1.w);
    a[8]  = f2bf(a2.x); a[9]  = f2bf(a2.y); a[10] = f2bf(a2.z); a[11] = f2bf(a2.w);
    a[12] = f2bf(a3.x); a[13] = f2bf(a3.y); a[14] = f2bf(a3.z); a[15] = f2bf(a3.w);
#pragma unroll
    for (int tn = 0; tn < NT; ++tn) {
      const v16bf b =
          *(const v16bf*)(Bp + ((size_t)(ks32 * NT + tn) * 32 + lane) * 16);
      acc[tn] = __builtin_amdgcn_wmma_f32_16x16x32_bf16(
          /*neg_a=*/false, a, /*neg_b=*/false, b,
          /*c_mod=*/(short)0, acc[tn], /*reuse_a=*/false, /*reuse_b=*/false);
    }
  }

#pragma unroll
  for (int tn = 0; tn < NT; ++tn) {
#pragma unroll
    for (int r = 0; r < 8; ++r) {
      const int m = wave * 16 + r + 8 * half;
      H[(size_t)m * cols + tn * 16 + lm] = acc[tn][r];
    }
  }
}

// One thread per (edge, 4-feature group): 1x b128 gather + 4 native f32 atomics.
__global__ void gcn_scatter4(const float* __restrict__ h,
                             const float* __restrict__ dinv,
                             const int* __restrict__ src,
                             const int* __restrict__ dst,
                             float* __restrict__ agg,
                             int e, int dshift) {
  const int gshift = dshift - 2;
  long long idx = (long long)blockIdx.x * blockDim.x + threadIdx.x;
  const long long total = (long long)e << gshift;
  if (idx >= total) return;
  const int ed = (int)(idx >> gshift);
  const int f  = (int)((idx & ((1 << gshift) - 1)) << 2);
  const int s = src[ed];
  const int t = dst[ed];
  const float w = dinv[s] * dinv[t];
  const float4 hv = *(const float4*)(h + ((size_t)s << dshift) + f);
  float* ap = agg + ((size_t)t << dshift) + f;
  atomic_add_f32(ap + 0, hv.x * w);
  atomic_add_f32(ap + 1, hv.y * w);
  atomic_add_f32(ap + 2, hv.z * w);
  atomic_add_f32(ap + 3, hv.w * w);
}

// out = relu(agg + h*dinv^2 + bias); out may alias agg. float4 per thread.
__global__ void gcn_finalize4(const float* __restrict__ agg,
                              const float* __restrict__ h,
                              const float* __restrict__ dinv,
                              const float* __restrict__ bias,
                              float* __restrict__ out,
                              int n, int dshift) {
  const int gshift = dshift - 2;
  long long idx = (long long)blockIdx.x * blockDim.x + threadIdx.x;
  const long long total = (long long)n << gshift;
  if (idx >= total) return;
  const int i = (int)(idx >> gshift);
  const int f = (int)((idx & ((1 << gshift) - 1)) << 2);
  const float di = dinv[i];
  const float d2 = di * di;
  const size_t base = ((size_t)i << dshift) + f;
  const float4 av = *(const float4*)(agg + base);
  const float4 hv = *(const float4*)(h + base);
  const float4 bv = *(const float4*)(bias + f);
  float4 o;
  o.x = av.x + hv.x * d2 + bv.x;
  o.y = av.y + hv.y * d2 + bv.y;
  o.z = av.z + hv.z * d2 + bv.z;
  o.w = av.w + hv.w * d2 + bv.w;
  o.x = o.x > 0.f ? o.x : 0.f;
  o.y = o.y > 0.f ? o.y : 0.f;
  o.z = o.z > 0.f ? o.z : 0.f;
  o.w = o.w > 0.f ? o.w : 0.f;
  *(float4*)(out + base) = o;
}

static inline int blocks_for(long long n, int bs) {
  return (int)((n + bs - 1) / bs);
}

extern "C" void kernel_launch(void* const* d_in, const int* in_sizes, int n_in,
                              void* d_out, int out_size, void* d_ws, size_t ws_size,
                              hipStream_t stream) {
  const float* x  = (const float*)d_in[0];
  const int*   ei = (const int*)d_in[1];
  const float* W1 = (const float*)d_in[2];
  const float* b1 = (const float*)d_in[3];
  const float* W2 = (const float*)d_in[4];
  const float* b2 = (const float*)d_in[5];
  const float* W3 = (const float*)d_in[6];
  const float* b3 = (const float*)d_in[7];

  const int N = in_sizes[0] / 128;   // 100000
  const int E = in_sizes[1] / 2;     // 1600000
  const int* src = ei;
  const int* dst = ei + E;

  // workspace layout (floats): dinv | h (N*128) | agg (N*128) | packed W (bf16)
  float*  dinv  = (float*)d_ws;
  float*  hbuf  = dinv + (1 << 17);            // 131072-float slot for dinv
  float*  agg   = hbuf + (size_t)N * 128;
  __bf16* bpack = (__bf16*)(agg + (size_t)N * 128);

  const int mtiles = N / 16;                   // 6250

  // --- normalization: deg -> dinv -------------------------------------------
  gcn_zero4<<<blocks_for((N + 3) / 4, 256), 256, 0, stream>>>((float4*)dinv, (N + 3) / 4);
  gcn_degree<<<blocks_for(E, 256), 256, 0, stream>>>(dst, dinv, E);
  gcn_rsqrt<<<blocks_for(N, 256), 256, 0, stream>>>(dinv, N);

  // --- layer 1: x(N,128) @ W1(128,128) --------------------------------------
  {
    const int cols = 128, dshift = 7;
    gcn_pack_w<<<blocks_for(cols * 128, 256), 256, 0, stream>>>(W1, bpack, cols / 16);
    gcn_gemm_wmma<8><<<blocks_for((long long)mtiles * 32, 256), 256, 0, stream>>>(x, bpack, hbuf, mtiles);
    const long long n4 = ((long long)N << dshift) >> 2;
    gcn_zero4<<<blocks_for(n4, 256), 256, 0, stream>>>((float4*)agg, n4);
    gcn_scatter4<<<blocks_for((long long)E << (dshift - 2), 256), 256, 0, stream>>>(hbuf, dinv, src, dst, agg, E, dshift);
    gcn_finalize4<<<blocks_for((long long)N << (dshift - 2), 256), 256, 0, stream>>>(agg, hbuf, dinv, b1, agg, N, dshift);
  }

  // --- layer 2: agg(N,128) @ W2(128,128) ------------------------------------
  {
    const int cols = 128, dshift = 7;
    gcn_pack_w<<<blocks_for(cols * 128, 256), 256, 0, stream>>>(W2, bpack, cols / 16);
    gcn_gemm_wmma<8><<<blocks_for((long long)mtiles * 32, 256), 256, 0, stream>>>(agg, bpack, hbuf, mtiles);
    const long long n4 = ((long long)N << dshift) >> 2;
    gcn_zero4<<<blocks_for(n4, 256), 256, 0, stream>>>((float4*)agg, n4);
    gcn_scatter4<<<blocks_for((long long)E << (dshift - 2), 256), 256, 0, stream>>>(hbuf, dinv, src, dst, agg, E, dshift);
    gcn_finalize4<<<blocks_for((long long)N << (dshift - 2), 256), 256, 0, stream>>>(agg, hbuf, dinv, b2, agg, N, dshift);
  }

  // --- layer 3: agg(N,128) @ W3(128,64) -> d_out ----------------------------
  {
    const int cols = 64, dshift = 6;
    gcn_pack_w<<<blocks_for(cols * 128, 256), 256, 0, stream>>>(W3, bpack, cols / 16);
    gcn_gemm_wmma<4><<<blocks_for((long long)mtiles * 32, 256), 256, 0, stream>>>(agg, bpack, hbuf, mtiles);
    const long long n4 = ((long long)N << dshift) >> 2;
    gcn_zero4<<<blocks_for(n4, 256), 256, 0, stream>>>((float4*)agg, n4);
    gcn_scatter4<<<blocks_for((long long)E << (dshift - 2), 256), 256, 0, stream>>>(hbuf, dinv, src, dst, agg, E, dshift);
    gcn_finalize4<<<blocks_for((long long)N << (dshift - 2), 256), 256, 0, stream>>>(agg, hbuf, dinv, b3, (float*)d_out, N, dshift);
  }
}
